// FSMRE_67800353734746
// MI455X (gfx1250) — matrix operand order repaired and checked by reading, not compiled, not run on hardware
//
#include <hip/hip_runtime.h>
#include <hip/hip_bf16.h>
#include <math.h>

// Problem constants
#define S_ 32
#define E_ 64
#define D_ 256
#define L_ 5
#define N_ 5   // proto count (== L)

typedef __attribute__((ext_vector_type(16))) __bf16 v16bf;
typedef __attribute__((ext_vector_type(8)))  __bf16 v8bf;
typedef __attribute__((ext_vector_type(8)))  float  v8f;
typedef __attribute__((ext_vector_type(4)))  unsigned int u32x4;
typedef __attribute__((ext_vector_type(8)))  int i32x8;
typedef __attribute__((ext_vector_type(4)))  int i32x4;

// ---------------- workspace layout (float units) ----------------
#define OFF_PA     0                      // 5*256
#define OFF_PB     1280                   // 5*256
#define OFF_CTX    2560                   // 5*256
#define OFF_CC2    3840                   // 5
#define OFF_PPA    3845                   // 5
#define OFF_PPB    3850                   // 5
#define OFF_C      3855                   // 5
#define OFF_XW     4096                   // 2048*256 f32
#define OFF_XSELF  528384                 // 2048*256 f32
#define OFF_WBUF   1052672                // 32*64*64*5 f32 (layout [s][v][k][u])
#define OFF_DA     3674112                // 32*5*64*5
#define OFF_DB     3725312                // 32*5*64*5
#define OFF_XB     3776512                // 2048*256 bf16 (= 262144 floats)
#define OFF_WT     4038656                // 256*256 bf16 transposed (= 32768 floats)
#define OFF_WST    4071424                // 256*256 bf16 transposed (= 32768 floats)

// ---------------- K1: prototypes, centers, norms, prior ----------------
__global__ __launch_bounds__(256) void k_precompute(
    const float* __restrict__ sup_entities,  // (5,5,512)
    const float* __restrict__ sup_context,   // (5,5,256)
    const int*   __restrict__ counts,        // (5,)
    float* __restrict__ pa, float* __restrict__ pb, float* __restrict__ ctx,
    float* __restrict__ cc2, float* __restrict__ ppa, float* __restrict__ ppb,
    float* __restrict__ cvec) {
  int tid = threadIdx.x;
  for (int i = tid; i < L_ * 2 * D_; i += 256) {
    int n = i / (2 * D_), d = i % (2 * D_);
    float s = 0.f;
    #pragma unroll
    for (int j = 0; j < 5; ++j) s += sup_entities[(n * 5 + j) * (2 * D_) + d];
    s *= 0.2f;
    if (d < D_) pa[n * D_ + d] = s;
    else        pb[n * D_ + (d - D_)] = s;
  }
  for (int i = tid; i < L_ * D_; i += 256) {
    int k = i / D_, d = i % D_;
    float s = 0.f;
    #pragma unroll
    for (int j = 0; j < 5; ++j) s += sup_context[(k * 5 + j) * D_ + d];
    ctx[i] = s * 0.2f;
  }
  __syncthreads();
  if (tid < L_) {
    float s0 = 0.f, s1 = 0.f, s2 = 0.f;
    for (int d = 0; d < D_; ++d) {
      float a = ctx[tid * D_ + d]; s0 += a * a;
      float b = pa[tid * D_ + d];  s1 += b * b;
      float c = pb[tid * D_ + d];  s2 += c * c;
    }
    cc2[tid] = s0; ppa[tid] = s1; ppb[tid] = s2;
  }
  if (tid == 0) {
    float cnt[L_], total = 0.f;
    #pragma unroll
    for (int n = 0; n < L_; ++n) { cnt[n] = (float)counts[n] + 1.0f; total += cnt[n]; }
    #pragma unroll
    for (int n = 0; n < L_; ++n) cvec[n] = cnt[n] / (total - cnt[n]);
  }
}

// ---------------- K1b: f32 -> bf16 conversion (row-major copy) --------------
__global__ __launch_bounds__(256) void k_to_bf16(const float* __restrict__ src,
                                                 __bf16* __restrict__ dst, int n) {
  int i = blockIdx.x * 256 + threadIdx.x;
  if (i < n) dst[i] = (__bf16)src[i];
}

// ---------------- K1c: f32 (K,N) -> bf16 (N,K) transpose-convert ------------
__global__ __launch_bounds__(256) void k_transpose_bf16(const float* __restrict__ src,
                                                        __bf16* __restrict__ dst,
                                                        int K, int N) {
  int i = blockIdx.x * 256 + threadIdx.x;
  if (i < K * N) {
    int k = i / N, n = i % N;
    dst[(size_t)n * K + k] = (__bf16)src[i];
  }
}

// ---------------- K2: C = A(MxK) * B(KxN) with Bt=(N,K), bf16 WMMA ----------
// one wave per 16x16 C tile; all operands loaded as b128 in exact WMMA layout
__global__ __launch_bounds__(32) void k_gemm_wmma(
    const __bf16* __restrict__ A,    // (M,256) row-major bf16
    const __bf16* __restrict__ Bt,   // (256,256) bf16, B transposed (n-major)
    float* __restrict__ C, int N) {
  const int tileN = blockIdx.x;
  const int tileM = blockIdx.y;
  const int lane  = threadIdx.x;     // 0..31, EXEC all ones (WMMA requirement)
  const int half  = lane >> 4;
  const int l16   = lane & 15;

  const __bf16* arow = A  + (size_t)(tileM * 16 + l16) * D_;
  const __bf16* brow = Bt + (size_t)(tileN * 16 + l16) * D_;

  v8f acc = {};
  #pragma unroll
  for (int kk = 0; kk < D_; kk += 32) {
    // A 16x32 layout: lane (m=l16, half) holds K in [8h,8h+8) and [16+8h,16+8h+8)
    v8bf a_lo = *(const v8bf*)(arow + kk + 8 * half);
    v8bf a_hi = *(const v8bf*)(arow + kk + 16 + 8 * half);
    v16bf a = __builtin_shufflevector(a_lo, a_hi,
        0, 1, 2, 3, 4, 5, 6, 7, 8, 9, 10, 11, 12, 13, 14, 15);
    // B 32x16 layout: lane (n=l16, half) holds K in [8h,8h+8) and [16+8h,16+8h+8)
    v8bf b_lo = *(const v8bf*)(brow + kk + 8 * half);
    v8bf b_hi = *(const v8bf*)(brow + kk + 16 + 8 * half);
    v16bf b = __builtin_shufflevector(b_lo, b_hi,
        0, 1, 2, 3, 4, 5, 6, 7, 8, 9, 10, 11, 12, 13, 14, 15);
    acc = __builtin_amdgcn_wmma_f32_16x16x32_bf16(
        /*neg_a=*/false, a, /*neg_b=*/false, b,
        /*c_mod=*/(short)0, acc, /*reuse_a=*/false, /*reuse_b=*/false);
  }
  // D layout: VGPR r -> row (r + 8*half), col l16
  #pragma unroll
  for (int r = 0; r < 8; ++r) {
    int m = tileM * 16 + r + 8 * half;
    C[(size_t)m * N + tileN * 16 + l16] = acc[r];
  }
}

// ---------------- K3: inverse-square-distance weights via TDM ----------------
// One wave per block; TDM pulls an 8-row x 1KB tile of `context` into LDS
// (2-D descriptor per ISA 8.3/8.4), s_wait_tensorcnt, then reduce from LDS.
// w stored as [s][v][k][u] (u-contiguous for K4).
__global__ __launch_bounds__(32) void k_weights_tdm(
    const float* __restrict__ context,     // (S,E,E,D)
    const float* __restrict__ ctx,         // (5,D)
    const float* __restrict__ cc2,         // (5,)
    float* __restrict__ wbuf) {
  __shared__ float smem[8 * D_];           // 8 KB; only static LDS object -> offset 0
  const int lane = threadIdx.x;
  const int base = blockIdx.x * 8;         // first (s,u,v) row of this block

  unsigned long long ga =
      (unsigned long long)(uintptr_t)(context + (size_t)base * D_);

  // D# group 0: count=1, lds_addr=0, global_addr, type=2
  u32x4 g0;
  g0[0] = 1u;
  g0[1] = 0u;
  g0[2] = (unsigned)(ga & 0xffffffffull);
  g0[3] = (unsigned)((ga >> 32) & 0x1ffffffull) | (2u << 30);
  // D# group 1: data_size=4B (code 2); tensor_dim0=256; tensor_dim1=S*E*E;
  //             tile_dim0=256, tile_dim1=8; tensor_dim0_stride=256
  i32x8 g1;
  g1[0] = (int)(2u << 16);                       // data_size code 2 at [17:16]
  g1[1] = (int)(256u << 16);                     // tensor_dim0 lo16 at word1[31:16]
  g1[2] = 0;                                     // tensor_dim0 hi16 | tensor_dim1 lo16
  g1[3] = (int)((256u << 16) | 2u);              // tensor_dim1 hi16=2 | tile_dim0=256
  g1[4] = 8;                                     // tile_dim1=8, tile_dim2=0
  g1[5] = 256;                                   // tensor_dim0_stride lo32
  g1[6] = 0;
  g1[7] = 0;
  i32x4 z4 = {0, 0, 0, 0};
#if defined(__clang_major__) && (__clang_major__ >= 23)
  i32x8 z8 = {0, 0, 0, 0, 0, 0, 0, 0};
  __builtin_amdgcn_tensor_load_to_lds(g0, g1, z4, z4, z8, 0);
#else
  __builtin_amdgcn_tensor_load_to_lds(g0, g1, z4, z4, 0);
#endif
  __builtin_amdgcn_s_wait_tensorcnt(0);
  asm volatile("" ::: "memory");

  for (int r = 0; r < 8; ++r) {
    int idx = base + r;                    // s*E*E + u*E + v
    int v  = idx & 63;
    int su = idx >> 6;
    int u  = su & 63;
    int s  = su >> 6;

    float cn2 = 0.f, cr0 = 0.f, cr1 = 0.f, cr2 = 0.f, cr3 = 0.f, cr4 = 0.f;
    for (int d = lane; d < D_; d += 32) {
      float cd = smem[r * D_ + d];
      cn2 += cd * cd;
      cr0 += cd * ctx[0 * D_ + d];
      cr1 += cd * ctx[1 * D_ + d];
      cr2 += cd * ctx[2 * D_ + d];
      cr3 += cd * ctx[3 * D_ + d];
      cr4 += cd * ctx[4 * D_ + d];
    }
    #pragma unroll
    for (int off = 16; off > 0; off >>= 1) {
      cn2 += __shfl_xor(cn2, off, 32);
      cr0 += __shfl_xor(cr0, off, 32);
      cr1 += __shfl_xor(cr1, off, 32);
      cr2 += __shfl_xor(cr2, off, 32);
      cr3 += __shfl_xor(cr3, off, 32);
      cr4 += __shfl_xor(cr4, off, 32);
    }
    if (lane == 0) {
      float cr[5] = {cr0, cr1, cr2, cr3, cr4};
      float* wrow = wbuf + ((size_t)(s * E_ + v) * L_) * E_;
      #pragma unroll
      for (int k = 0; k < L_; ++k) {
        float d2 = cn2 - 2.0f * cr[k] + cc2[k];
        float w  = (u == v) ? 0.0f : 1.0f / fmaxf(d2, 1e-8f);
        wrow[(size_t)k * E_ + u] = w;
      }
    }
  }
}

// ---------------- K4: fused out-vector + squared distances ----------------
__global__ __launch_bounds__(256) void k_outdist(
    const float* __restrict__ xW,    // (S*E, D)
    const float* __restrict__ xself, // (S*E, D)
    const float* __restrict__ wbuf,  // [s][v][k][u]
    const float* __restrict__ pa, const float* __restrict__ pb,
    const float* __restrict__ ppa, const float* __restrict__ ppb,
    float* __restrict__ da, float* __restrict__ db) {
  int v  = blockIdx.x & 63;
  int sk = blockIdx.x >> 6;
  int k  = sk % L_;
  int s  = sk / L_;
  int d  = threadIdx.x;            // 0..255

  const float* wrow = wbuf + ((size_t)(s * E_ + v) * L_ + k) * E_;
  const float* xWs  = xW + (size_t)s * E_ * D_ + d;
  float o = xself[((size_t)s * E_ + v) * D_ + d];
  for (int u = 0; u < E_; ++u) o += wrow[u] * xWs[(size_t)u * D_];

  float vals[11];
  vals[0] = o * o;
  #pragma unroll
  for (int n = 0; n < N_; ++n) {
    vals[1 + n] = o * pa[n * D_ + d];
    vals[6 + n] = o * pb[n * D_ + d];
  }

  __shared__ float red[11 * 8];
  int lane = d & 31, wid = d >> 5;
  #pragma unroll
  for (int q = 0; q < 11; ++q) {
    float t = vals[q];
    #pragma unroll
    for (int off = 16; off > 0; off >>= 1) t += __shfl_xor(t, off, 32);
    if (lane == 0) red[q * 8 + wid] = t;
  }
  __syncthreads();
  if (threadIdx.x == 0) {
    float sums[11];
    #pragma unroll
    for (int q = 0; q < 11; ++q) {
      float t = 0.f;
      #pragma unroll
      for (int i = 0; i < 8; ++i) t += red[q * 8 + i];
      sums[q] = t;
    }
    size_t base = ((size_t)(s * L_ + k) * E_ + v) * N_;
    #pragma unroll
    for (int n = 0; n < N_; ++n) {
      da[base + n] = sums[0] - 2.0f * sums[1 + n] + ppa[n];
      db[base + n] = sums[0] - 2.0f * sums[6 + n] + ppb[n];
    }
  }
}

// ---------------- K5: pairwise logits + softmax over n, transposed store ------
__global__ __launch_bounds__(256) void k_softmax(
    const float* __restrict__ da, const float* __restrict__ db,
    const float* __restrict__ cvec, float* __restrict__ out) {
  long t = (long)blockIdx.x * 256 + threadIdx.x;   // (s,k,v1,v2)
  if (t >= (long)S_ * L_ * E_ * E_) return;
  int v2 = (int)(t & 63); long r = t >> 6;
  int v1 = (int)(r & 63); r >>= 6;
  int k  = (int)(r % L_);
  int s  = (int)(r / L_);

  const float* dav = da + ((size_t)(s * L_ + k) * E_ + v1) * N_;
  const float* dbv = db + ((size_t)(s * L_ + k) * E_ + v2) * N_;
  float l[N_], m = -INFINITY;
  #pragma unroll
  for (int n = 0; n < N_; ++n) {
    l[n] = dav[n] + dbv[n] + cvec[n];
    m = fmaxf(m, l[n]);
  }
  float sum = 0.f;
  #pragma unroll
  for (int n = 0; n < N_; ++n) { l[n] = __expf(l[n] - m); sum += l[n]; }
  float inv = 1.0f / sum;
  float* op = out + ((((size_t)s * E_ + v1) * E_ + v2) * L_ + k) * N_;
  #pragma unroll
  for (int n = 0; n < N_; ++n) op[n] = l[n] * inv;
}

extern "C" void kernel_launch(void* const* d_in, const int* in_sizes, int n_in,
                              void* d_out, int out_size, void* d_ws, size_t ws_size,
                              hipStream_t stream) {
  const float* sup_entities = (const float*)d_in[0];
  const float* sup_context  = (const float*)d_in[1];
  const float* x            = (const float*)d_in[2];
  const float* context      = (const float*)d_in[3];
  const float* W            = (const float*)d_in[4];
  const float* Wself        = (const float*)d_in[5];
  const int*   counts       = (const int*)d_in[6];
  float* out = (float*)d_out;
  float* ws  = (float*)d_ws;

  float*  pa    = ws + OFF_PA;
  float*  pb    = ws + OFF_PB;
  float*  ctx   = ws + OFF_CTX;
  float*  cc2   = ws + OFF_CC2;
  float*  ppa   = ws + OFF_PPA;
  float*  ppb   = ws + OFF_PPB;
  float*  cvec  = ws + OFF_C;
  float*  xWbuf = ws + OFF_XW;
  float*  xself = ws + OFF_XSELF;
  float*  wbuf  = ws + OFF_WBUF;
  float*  da    = ws + OFF_DA;
  float*  db    = ws + OFF_DB;
  __bf16* xb    = (__bf16*)(ws + OFF_XB);
  __bf16* Wt    = (__bf16*)(ws + OFF_WT);
  __bf16* Wst   = (__bf16*)(ws + OFF_WST);

  // K1: tiny precompute
  k_precompute<<<1, 256, 0, stream>>>(sup_entities, sup_context, counts,
                                      pa, pb, ctx, cc2, ppa, ppb, cvec);

  // K1b/K1c: one-time bf16 staging (x row-major; W/Wself transposed for B layout)
  k_to_bf16<<<(S_ * E_ * D_) / 256, 256, 0, stream>>>(x, xb, S_ * E_ * D_);
  k_transpose_bf16<<<(D_ * D_) / 256, 256, 0, stream>>>(W,     Wt,  D_, D_);
  k_transpose_bf16<<<(D_ * D_) / 256, 256, 0, stream>>>(Wself, Wst, D_, D_);

  // K2: x@W and x@Wself (M=2048, N=256, K=256) via WMMA bf16, b128 operand loads
  dim3 g2(D_ / 16, (S_ * E_) / 16);     // (16, 128) tiles
  k_gemm_wmma<<<g2, 32, 0, stream>>>(xb, Wt,  xWbuf, D_);
  k_gemm_wmma<<<g2, 32, 0, stream>>>(xb, Wst, xself, D_);

  // K3: weights from context stream (134 MB, read once) staged via TDM->LDS
  k_weights_tdm<<<(S_ * E_ * E_) / 8, 32, 0, stream>>>(context, ctx, cc2, wbuf);

  // K4: fused out-vector + distances (out tensor never hits memory)
  k_outdist<<<S_ * L_ * E_, 256, 0, stream>>>(xWbuf, xself, wbuf,
                                              pa, pb, ppa, ppb, da, db);

  // K5: pairwise logits + softmax, transposed store
  long rows = (long)S_ * L_ * E_ * E_;
  k_softmax<<<(int)((rows + 255) / 256), 256, 0, stream>>>(da, db, cvec, out);
}
